// Involution2d_17729624998532
// MI455X (gfx1250) — matrix operand order repaired and checked by reading, not compiled, not run on hardware
//
#include <hip/hip_runtime.h>
#include <hip/hip_bf16.h>

typedef unsigned short u16t;
typedef unsigned int   u32t;

typedef __attribute__((ext_vector_type(16))) __bf16 v16bf;
typedef __attribute__((ext_vector_type(8)))  float  v8f;

// fp32 -> bf16 round-to-nearest-even
__device__ __forceinline__ u16t f2bf(float f) {
    u32t u = __builtin_bit_cast(u32t, f);
    u += 0x7FFFu + ((u >> 16) & 1u);
    return (u16t)(u >> 16);
}

union Frag16 {
    v16bf v;
    u16t  u[16];
    uint4 q[2];
};

// ---------------------------------------------------------------------------
// Problem constants
// ---------------------------------------------------------------------------
#define Bn   8
#define Cn   256
#define Hn   56
#define Wn   56
#define HWn  (Hn * Wn)        // 3136
#define Gn   16
#define CGn  (Cn / Gn)        // 16
#define Kk   7
#define KKn  (Kk * Kk)        // 49
#define Crn  64               // C / 4
#define KKG  (KKn * Gn)       // 784

// LDS strides (padded for bank-conflict-free access)
#define K1_STRIDE   72        // u16 elems per ker1 row (144 B; 36*l mod 64 distinct for l<16)
#define KER_STRIDE  788       // f32 elems per kers row (788 mod 64 == 20; 20*p mod 64 distinct)

// Workspace layout (bytes) — only packed weights now
#define WRP_OFF   0           // packed w_reduce frags: 4*8*32*16 u16 = 32768 B
#define WSP_OFF   32768       // packed w_span   frags: 49*2*32*16 u16 = 100352 B

// ---------------------------------------------------------------------------
// Kernel 0: pack weights (fp32 -> bf16) into per-lane WMMA B-fragment order.
// B-frag layout (32x16 bf16): lane n<16 -> col n, K = 0..7 & 16..23;  lane>=16 -> col n-16, K = 8..15 & 24..31
// ---------------------------------------------------------------------------
__global__ void pack_weights_kernel(const float* __restrict__ w_reduce,
                                    const float* __restrict__ w_span,
                                    u16t* __restrict__ wrPack,
                                    u16t* __restrict__ wsPack) {
    int i = blockIdx.x * 256 + threadIdx.x;
    if (i < 4 * 8 * 32 * 16) {                 // w_reduce: nt[0,4) kc[0,8)
        int e    = i & 15;
        int lane = (i >> 4) & 31;
        int kc   = (i >> 9) & 7;
        int nt   = i >> 12;
        int n = nt * 16 + (lane & 15);
        int K = kc * 32 + ((e < 8) ? e : 8 + e) + ((lane < 16) ? 0 : 8);
        wrPack[i] = f2bf(w_reduce[n * Cn + K]);
    }
    if (i < 49 * 2 * 32 * 16) {                // w_span: nt[0,49) kc[0,2)
        int e    = i & 15;
        int lane = (i >> 4) & 31;
        int kc   = (i >> 9) & 1;
        int nt   = i >> 10;
        int n = nt * 16 + (lane & 15);
        int K = kc * 32 + ((e < 8) ? e : 8 + e) + ((lane < 16) ? 0 : 8);
        wsPack[i] = f2bf(w_span[n * Crn + K]);
    }
}

// ---------------------------------------------------------------------------
// Fused kernel: per (16-pixel tile, batch) workgroup of 256 threads (8 waves).
//   Phase 0 (waves 0-3): ker1 tile [16 x 64] = x-tile * w_reduce^T + b_reduce (WMMA, bf16 -> LDS)
//   Phase 1 (all waves): kers tile [16 x 784] = ker1 * w_span^T + b_span     (WMMA, fp32 -> LDS)
//   Phase 2 (16 px x 16 ch): 49-tap involution gather, ker from LDS, x from global.
// The 79 MB ker intermediate never touches HBM.
// ---------------------------------------------------------------------------
__global__ void involution_fused_kernel(const float* __restrict__ x,
                                        const u16t* __restrict__ wrPack,
                                        const u16t* __restrict__ wsPack,
                                        const float* __restrict__ b_reduce,
                                        const float* __restrict__ b_span,
                                        float* __restrict__ out) {
    __shared__ u16t  k1s[16 * K1_STRIDE];            //  2304 B  (ker1 tile, bf16)
    __shared__ float kers[16 * KER_STRIDE];          // 50432 B  (span tile, fp32)

    const int ptile = blockIdx.x;                    // 16-pixel tile
    const int b     = blockIdx.y;
    const int t     = threadIdx.x;
    const int wave  = t >> 5;                        // 0..7
    const int lane  = t & 31;
    const int nlo   = lane & 15;
    const bool hi   = lane >= 16;
    const int hw0   = ptile * 16;

    const float* xb = x + (size_t)b * Cn * HWn;

    // ---------------- Phase 0: reduce GEMM (waves 0-3, one 16x16 N-tile each) --
    if (wave < 4) {
        const int nt = wave;                         // N-tile of the 64 outputs
        v8f acc;
        {
            float bias = b_reduce[nt * 16 + nlo];
            for (int r = 0; r < 8; ++r) acc[r] = bias;
        }
        const int m = hw0 + nlo;                     // A row this lane supplies
        for (int kc = 0; kc < 8; ++kc) {
            // A frag (16x32 bf16): lanes<16 -> K 0..7 & 16..23 ; lanes>=16 -> 8..15 & 24..31
            const int kb_lo = kc * 32 + (hi ? 8 : 0);
            const int kb_hi = kc * 32 + (hi ? 24 : 16);
            Frag16 a;
            #pragma unroll
            for (int e = 0; e < 8; ++e)
                a.u[e] = f2bf(xb[(size_t)(kb_lo + e) * HWn + m]);
            #pragma unroll
            for (int e = 0; e < 8; ++e)
                a.u[8 + e] = f2bf(xb[(size_t)(kb_hi + e) * HWn + m]);

            Frag16 bfr;
            const uint4* bp = (const uint4*)(wrPack + (((nt * 8) + kc) * 32 + lane) * 16);
            bfr.q[0] = bp[0];
            bfr.q[1] = bp[1];
            acc = __builtin_amdgcn_wmma_f32_16x16x32_bf16(
                false, a.v, false, bfr.v, (short)0, acc, false, false);
        }
        // D: VGPR r -> M = r (lanes<16) / 8+r ; N = lane&15.  Store bf16 to LDS.
        #pragma unroll
        for (int r = 0; r < 8; ++r) {
            int mr = hi ? 8 + r : r;
            k1s[mr * K1_STRIDE + nt * 16 + nlo] = f2bf(acc[r]);
        }
    }
    __syncthreads();

    // ---------------- Phase 1: span GEMM (all 8 waves, 49 N-tiles round-robin) --
    {
        // A fragments (shared across all N-tiles): 2 K-chunks from LDS ker1 tile
        Frag16 a0, a1;
        {
            const u16t* row = &k1s[nlo * K1_STRIDE];
            const int kb0_lo = hi ? 8 : 0,  kb0_hi = hi ? 24 : 16;
            a0.q[0] = *(const uint4*)(row + kb0_lo);
            a0.q[1] = *(const uint4*)(row + kb0_hi);
            a1.q[0] = *(const uint4*)(row + 32 + kb0_lo);
            a1.q[1] = *(const uint4*)(row + 32 + kb0_hi);
        }
        for (int nt = wave; nt < 49; nt += 8) {
            v8f acc;
            {
                float bias = b_span[nt * 16 + nlo];
                for (int r = 0; r < 8; ++r) acc[r] = bias;
            }
            const uint4* bp0 = (const uint4*)(wsPack + ((nt * 2 + 0) * 32 + lane) * 16);
            const uint4* bp1 = (const uint4*)(wsPack + ((nt * 2 + 1) * 32 + lane) * 16);
            Frag16 bfr;
            bfr.q[0] = bp0[0]; bfr.q[1] = bp0[1];
            acc = __builtin_amdgcn_wmma_f32_16x16x32_bf16(
                false, a0.v, false, bfr.v, (short)0, acc, false, false);
            bfr.q[0] = bp1[0]; bfr.q[1] = bp1[1];
            acc = __builtin_amdgcn_wmma_f32_16x16x32_bf16(
                false, a1.v, false, bfr.v, (short)0, acc, false, false);
            #pragma unroll
            for (int r = 0; r < 8; ++r) {
                int mr = hi ? 8 + r : r;                       // pixel within tile
                kers[mr * KER_STRIDE + nt * 16 + nlo] = acc[r];
            }
        }
    }
    __syncthreads();

    // ---------------- Phase 2: involution gather (16 pixels x 16 channels) ------
    {
        const int p  = t & 15;            // pixel (fast dim -> coalesced x/out)
        const int cp = t >> 4;            // channel within group
        const int hw = hw0 + p;
        const int h  = hw / Wn;
        const int w  = hw - h * Wn;
        const float* kp = &kers[p * KER_STRIDE];

        for (int g = 0; g < Gn; ++g) {
            const int c = g * CGn + cp;
            const float* xc = xb + (size_t)c * HWn;
            if (g + 1 < Gn) {             // prefetch next group's channel plane row
                __builtin_prefetch((const void*)(xb + (size_t)(c + CGn) * HWn + hw), 0, 1);
            }
            float accv = 0.0f;
            const float* kg = kp + g * KKn;
            #pragma unroll
            for (int kh = 0; kh < Kk; ++kh) {
                const int hh = h + kh - 3;
                if (hh < 0 || hh >= Hn) continue;
                #pragma unroll
                for (int kw = 0; kw < Kk; ++kw) {
                    const int ww = w + kw - 3;
                    if (ww < 0 || ww >= Wn) continue;
                    accv = fmaf(kg[kh * Kk + kw], xc[hh * Wn + ww], accv);
                }
            }
            out[((size_t)b * Cn + c) * HWn + hw] = accv;
        }
    }
}

// ---------------------------------------------------------------------------
extern "C" void kernel_launch(void* const* d_in, const int* in_sizes, int n_in,
                              void* d_out, int out_size, void* d_ws, size_t ws_size,
                              hipStream_t stream) {
    (void)in_sizes; (void)n_in; (void)out_size; (void)ws_size;

    const float* x        = (const float*)d_in[0];
    const float* w_reduce = (const float*)d_in[1];
    const float* b_reduce = (const float*)d_in[2];
    const float* w_span   = (const float*)d_in[3];
    const float* b_span   = (const float*)d_in[4];
    float*       out      = (float*)d_out;

    char* ws = (char*)d_ws;
    u16t* wrPack = (u16t*)(ws + WRP_OFF);
    u16t* wsPack = (u16t*)(ws + WSP_OFF);

    pack_weights_kernel<<<196, 256, 0, stream>>>(w_reduce, w_span, wrPack, wsPack);
    involution_fused_kernel<<<dim3(HWn / 16, Bn), 256, 0, stream>>>(
        x, wrPack, wsPack, b_reduce, b_span, out);
}
